// MoE_20624432955731
// MI455X (gfx1250) — matrix operand order repaired and checked by reading, not compiled, not run on hardware
//
#include <hip/hip_runtime.h>
#include <hip/hip_bf16.h>

typedef unsigned short ushort_t;
typedef __attribute__((ext_vector_type(16))) __bf16 v16bf;
typedef __attribute__((ext_vector_type(8)))  float  v8f;

#define NTOK   8192
#define DMODEL 1024
#define HDIM   2048
#define NEXP   8
#define TOPK   2
#define CAP    2560

__device__ __forceinline__ ushort_t f2bf(float f) {
  unsigned int u = __float_as_uint(f);
  unsigned int r = (u + 0x7FFFu + ((u >> 16) & 1u)) >> 16;  // round-to-nearest-even
  return (ushort_t)r;
}

union FragBF { v16bf v; uint4 q[2]; };

// Async global->LDS staging (ASYNCcnt-tracked). INST_OFFSET is added to BOTH
// the LDS and global addresses per the ISA pseudocode.
__device__ __forceinline__ void async_copy32(unsigned lds_off, const ushort_t* gptr) {
  asm volatile(
      "global_load_async_to_lds_b128 %0, %1, off\n\t"
      "global_load_async_to_lds_b128 %0, %1, off offset:16"
      :: "v"(lds_off), "v"((unsigned long long)gptr) : "memory");
}
__device__ __forceinline__ void async_copy64(unsigned lds_off, const ushort_t* gptr) {
  asm volatile(
      "global_load_async_to_lds_b128 %0, %1, off\n\t"
      "global_load_async_to_lds_b128 %0, %1, off offset:16\n\t"
      "global_load_async_to_lds_b128 %0, %1, off offset:32\n\t"
      "global_load_async_to_lds_b128 %0, %1, off offset:48"
      :: "v"(lds_off), "v"((unsigned long long)gptr) : "memory");
}
// Async loads complete in order: waiting <=6 after issuing the next tile's 6
// loads guarantees the previous tile's 6 loads have landed.
__device__ __forceinline__ void wait_async6() {
  asm volatile("s_wait_asynccnt 0x6" ::: "memory");
}
__device__ __forceinline__ void wait_async0() {
  asm volatile("s_wait_asynccnt 0x0" ::: "memory");
}

// One double-buffered pipeline step: prefetch next tile, consume current tile.
// B-fragments (4) are held; A-fragments are streamed one at a time to keep
// VGPR pressure under 256 (128 acc + 32 b + 8 a + addressing).
__device__ __forceinline__ void gemm_step(
    const ushort_t* __restrict__ AsC, const ushort_t* __restrict__ BsC,
    unsigned lA_nxt, unsigned lB_nxt,
    const ushort_t* gA_nxt, const ushort_t* gB_nxt, bool has_next,
    int wm, int wn, int hsel, int mr, v8f (&acc)[4][4]) {
  if (has_next) {
    async_copy32(lA_nxt, gA_nxt);
    async_copy64(lB_nxt, gB_nxt);
    wait_async6();
  } else {
    wait_async0();
  }
  __syncthreads();   // current tile visible from all waves

  FragBF b[4];
#pragma unroll
  for (int nt = 0; nt < 4; nt++) {
    const ushort_t* p = BsC + (wn * 64 + nt * 16 + mr) * 40 + hsel * 16;
    b[nt].q[0] = ((const uint4*)p)[0];
    b[nt].q[1] = ((const uint4*)p)[1];
  }
#pragma unroll
  for (int mt = 0; mt < 4; mt++) {
    FragBF a;
    const ushort_t* p = AsC + (wm * 64 + mt * 16 + mr) * 40;
    a.q[0] = *(const uint4*)(p + hsel * 16);
    a.q[1] = *(const uint4*)(p + 8 + hsel * 16);
#pragma unroll
    for (int nt = 0; nt < 4; nt++)
      acc[mt][nt] = __builtin_amdgcn_wmma_f32_16x16x32_bf16(
          false, a.v, false, b[nt].v, (short)0, acc[mt][nt], false, false);
  }
  __syncthreads();   // waits dscnt first: all reads of this buffer done -> reusable
}

// ---------------------------------------------------------------- zero accums
__global__ void zero_acc_kernel(float* acc) {
  if (threadIdx.x < 64) acc[threadIdx.x] = 0.0f;
}

// ---------------------------------------------------------------- gating
__global__ __launch_bounds__(256) void gating_kernel(
    const float* __restrict__ x, const float* __restrict__ Wg,
    const float* __restrict__ bg, int* __restrict__ tki,
    float* __restrict__ tkp, float* __restrict__ acc) {
  const int lane = threadIdx.x & 31;
  const int wave = threadIdx.x >> 5;
  const int n = blockIdx.x * 8 + wave;
  if (n >= NTOK) return;

  float s[8] = {0.f,0.f,0.f,0.f,0.f,0.f,0.f,0.f};
  const float* xr = x + (size_t)n * DMODEL;
  for (int d = lane; d < DMODEL; d += 32) {
    float xv = xr[d];
    const float4* wr = (const float4*)(Wg + d * NEXP);
    float4 w0 = wr[0], w1 = wr[1];
    s[0] += xv * w0.x; s[1] += xv * w0.y; s[2] += xv * w0.z; s[3] += xv * w0.w;
    s[4] += xv * w1.x; s[5] += xv * w1.y; s[6] += xv * w1.z; s[7] += xv * w1.w;
  }
#pragma unroll
  for (int o = 16; o > 0; o >>= 1) {
#pragma unroll
    for (int e = 0; e < 8; e++) s[e] += __shfl_down(s[e], o, 32);
  }
  if (lane == 0) {
    float l[8], mx = -1e30f;
#pragma unroll
    for (int e = 0; e < 8; e++) { l[e] = s[e] + bg[e]; mx = fmaxf(mx, l[e]); }
    float p[8], sum = 0.f;
#pragma unroll
    for (int e = 0; e < 8; e++) { p[e] = expf(l[e] - mx); sum += p[e]; }
    float inv = 1.0f / sum;
#pragma unroll
    for (int e = 0; e < 8; e++) p[e] *= inv;
    int i0 = 0;
#pragma unroll
    for (int e = 1; e < 8; e++) if (p[e] > p[i0]) i0 = e;
    int i1 = (i0 == 0) ? 1 : 0;
#pragma unroll
    for (int e = 0; e < 8; e++) if (e != i0 && p[e] > p[i1]) i1 = e;
    tki[2 * n]     = i0;  tki[2 * n + 1] = i1;
    tkp[2 * n]     = p[i0]; tkp[2 * n + 1] = p[i1];
#pragma unroll
    for (int e = 0; e < 8; e++) atomicAdd(&acc[e], p[e]);        // importance
    atomicAdd(&acc[8  + i0], 1.0f);  atomicAdd(&acc[8  + i1], 1.0f);   // usage
    atomicAdd(&acc[16 + i0], p[i0]); atomicAdd(&acc[16 + i1], p[i1]);  // routing
  }
}

// ---------------------------------------------------------------- exact first-come rank
__global__ void rank_kernel(const int* __restrict__ tki, int* __restrict__ rnk) {
  int e = threadIdx.x;
  if (e >= NEXP) return;
  int c = 0;
  for (int i = 0; i < NTOK * TOPK; i++) {
    if (tki[i] == e) rnk[i] = c++;
  }
}

// ---------------------------------------------------------------- dispatch x -> bf16 buf[E,C,D]
__global__ __launch_bounds__(256) void dispatch_kernel(
    const float* __restrict__ x, const int* __restrict__ tki,
    const int* __restrict__ rnk, ushort_t* __restrict__ buf) {
  const int slot = blockIdx.x;
  const int e = tki[slot];
  const int r = rnk[slot];
  if (r >= CAP) return;
  const int token = slot >> 1;
  const int c = threadIdx.x * 4;
  float4 v = *(const float4*)(x + (size_t)token * DMODEL + c);
  unsigned int lo = (unsigned int)f2bf(v.x) | ((unsigned int)f2bf(v.y) << 16);
  unsigned int hi = (unsigned int)f2bf(v.z) | ((unsigned int)f2bf(v.w) << 16);
  uint2 pk; pk.x = lo; pk.y = hi;
  *(uint2*)(buf + ((size_t)e * CAP + r) * DMODEL + c) = pk;
}

// ---------------------------------------------------------------- weight convert+transpose
__global__ __launch_bounds__(256) void wconv1_kernel(const float* __restrict__ W1,
                                                     ushort_t* __restrict__ W1t) {
  const int i = blockIdx.x * 256 + threadIdx.x;
  const int e = i >> 21;               // D*H = 2^21
  const int rem = i & ((1 << 21) - 1);
  const int d = rem >> 11;             // H = 2^11
  const int h = rem & (HDIM - 1);
  W1t[((size_t)e * HDIM + h) * DMODEL + d] = f2bf(W1[i]);
}
__global__ __launch_bounds__(256) void wconv2_kernel(const float* __restrict__ W2,
                                                     ushort_t* __restrict__ W2t) {
  const int i = blockIdx.x * 256 + threadIdx.x;
  const int e = i >> 22;               // H*H = 2^22
  const int rem = i & ((1 << 22) - 1);
  const int hin = rem >> 11;
  const int o = rem & (HDIM - 1);
  W2t[((size_t)e * HDIM + o) * HDIM + hin] = f2bf(W2[i]);
}

// ================================================================ WMMA GEMMs
// Block tile 128(M) x 256(N), k-step 32, 8 waves in 2x4, wave tile 64x64.
// Double-buffered async global->LDS staging; manual unroll-by-2 keeps LDS
// addresses compile-time constant (S is even for both GEMMs).

// GEMM1: h = gelu(buf @ W1 + b1) -> bf16
__global__ __launch_bounds__(256) void gemm1_kernel(
    const ushort_t* __restrict__ Abuf, const ushort_t* __restrict__ Bt,
    const float* __restrict__ b1, ushort_t* __restrict__ Hbuf) {
  const int e  = blockIdx.z;
  const int m0 = blockIdx.x * 128;
  const int n0 = blockIdx.y * 256;
  const int K  = DMODEL;
  const ushort_t* A = Abuf + (size_t)e * CAP * DMODEL;
  const ushort_t* B = Bt  + (size_t)e * HDIM * DMODEL;
  __shared__ __align__(16) ushort_t As[2][128 * 40];
  __shared__ __align__(16) ushort_t Bs[2][256 * 40];
  const int tid  = threadIdx.x;
  const int lane = tid & 31, wave = tid >> 5;
  const int wm = wave & 1, wn = wave >> 1;
  const int arow = tid & 127, apart = tid >> 7;
  const int hsel = lane >> 4, mr = lane & 15;

  const ushort_t* gA = A + (size_t)(m0 + arow) * K + apart * 16;
  const ushort_t* gB = B + (size_t)(n0 + tid) * K;
  const unsigned lA0 = (unsigned)(size_t)(As[0] + arow * 40 + apart * 16);
  const unsigned lA1 = (unsigned)(size_t)(As[1] + arow * 40 + apart * 16);
  const unsigned lB0 = (unsigned)(size_t)(Bs[0] + tid * 40);
  const unsigned lB1 = (unsigned)(size_t)(Bs[1] + tid * 40);

  v8f acc[4][4];
#pragma unroll
  for (int i = 0; i < 4; i++)
#pragma unroll
    for (int j = 0; j < 4; j++)
#pragma unroll
      for (int q = 0; q < 8; q++) acc[i][j][q] = 0.0f;

  const int S = K / 32;   // 32 (even)
  async_copy32(lA0, gA);
  async_copy64(lB0, gB);
  for (int i = 0; i < S; i += 2) {
    gemm_step(As[0], Bs[0], lA1, lB1, gA + (size_t)(i + 1) * 32,
              gB + (size_t)(i + 1) * 32, true, wm, wn, hsel, mr, acc);
    gemm_step(As[1], Bs[1], lA0, lB0, gA + (size_t)(i + 2) * 32,
              gB + (size_t)(i + 2) * 32, (i + 2) < S, wm, wn, hsel, mr, acc);
  }

#pragma unroll
  for (int mt = 0; mt < 4; mt++) {
#pragma unroll
    for (int nt = 0; nt < 4; nt++) {
      const int ng = n0 + wn * 64 + nt * 16 + mr;
      const float bias = b1[e * HDIM + ng];
#pragma unroll
      for (int j = 0; j < 8; j++) {
        const int mg = m0 + wm * 64 + mt * 16 + hsel * 8 + j;
        float v = acc[mt][nt][j] + bias;
        float g = 0.5f * v * (1.0f + erff(v * 0.70710678118654752f));
        Hbuf[((size_t)e * CAP + mg) * HDIM + ng] = f2bf(g);
      }
    }
  }
}

// GEMM2: y = h @ W2 + b2 -> f32
__global__ __launch_bounds__(256) void gemm2_kernel(
    const ushort_t* __restrict__ Hbuf, const ushort_t* __restrict__ Bt,
    const float* __restrict__ b2, float* __restrict__ Ybuf) {
  const int e  = blockIdx.z;
  const int m0 = blockIdx.x * 128;
  const int n0 = blockIdx.y * 256;
  const int K  = HDIM;
  const ushort_t* A = Hbuf + (size_t)e * CAP * HDIM;
  const ushort_t* B = Bt  + (size_t)e * HDIM * HDIM;
  __shared__ __align__(16) ushort_t As[2][128 * 40];
  __shared__ __align__(16) ushort_t Bs[2][256 * 40];
  const int tid  = threadIdx.x;
  const int lane = tid & 31, wave = tid >> 5;
  const int wm = wave & 1, wn = wave >> 1;
  const int arow = tid & 127, apart = tid >> 7;
  const int hsel = lane >> 4, mr = lane & 15;

  const ushort_t* gA = A + (size_t)(m0 + arow) * K + apart * 16;
  const ushort_t* gB = B + (size_t)(n0 + tid) * K;
  const unsigned lA0 = (unsigned)(size_t)(As[0] + arow * 40 + apart * 16);
  const unsigned lA1 = (unsigned)(size_t)(As[1] + arow * 40 + apart * 16);
  const unsigned lB0 = (unsigned)(size_t)(Bs[0] + tid * 40);
  const unsigned lB1 = (unsigned)(size_t)(Bs[1] + tid * 40);

  v8f acc[4][4];
#pragma unroll
  for (int i = 0; i < 4; i++)
#pragma unroll
    for (int j = 0; j < 4; j++)
#pragma unroll
      for (int q = 0; q < 8; q++) acc[i][j][q] = 0.0f;

  const int S = K / 32;   // 64 (even)
  async_copy32(lA0, gA);
  async_copy64(lB0, gB);
  for (int i = 0; i < S; i += 2) {
    gemm_step(As[0], Bs[0], lA1, lB1, gA + (size_t)(i + 1) * 32,
              gB + (size_t)(i + 1) * 32, true, wm, wn, hsel, mr, acc);
    gemm_step(As[1], Bs[1], lA0, lB0, gA + (size_t)(i + 2) * 32,
              gB + (size_t)(i + 2) * 32, (i + 2) < S, wm, wn, hsel, mr, acc);
  }

#pragma unroll
  for (int mt = 0; mt < 4; mt++) {
#pragma unroll
    for (int nt = 0; nt < 4; nt++) {
      const int ng = n0 + wn * 64 + nt * 16 + mr;
      const float bias = b2[e * HDIM + ng];
#pragma unroll
      for (int j = 0; j < 8; j++) {
        const int mg = m0 + wm * 64 + mt * 16 + hsel * 8 + j;
        Ybuf[((size_t)e * CAP + mg) * HDIM + ng] = acc[mt][nt][j] + bias;
      }
    }
  }
}

// ---------------------------------------------------------------- combine (no atomics: per-token)
__global__ __launch_bounds__(256) void combine_kernel(
    const float* __restrict__ Ybuf, const int* __restrict__ tki,
    const float* __restrict__ tkp, const int* __restrict__ rnk,
    float* __restrict__ out) {
  const int n = blockIdx.x;
  const int e0 = tki[2 * n], e1 = tki[2 * n + 1];
  const int r0 = rnk[2 * n], r1 = rnk[2 * n + 1];
  const float p0 = tkp[2 * n], p1 = tkp[2 * n + 1];
  const int c = threadIdx.x * 8;
  float o[8];
#pragma unroll
  for (int j = 0; j < 8; j++) o[j] = 0.0f;
  if (r0 < CAP) {
    const float* y = Ybuf + ((size_t)e0 * CAP + r0) * HDIM + c;
#pragma unroll
    for (int j = 0; j < 8; j++) o[j] += p0 * y[j];
  }
  if (r1 < CAP) {
    const float* y = Ybuf + ((size_t)e1 * CAP + r1) * HDIM + c;
#pragma unroll
    for (int j = 0; j < 8; j++) o[j] += p1 * y[j];
  }
  float4* dst = (float4*)(out + (size_t)n * HDIM + c);
  float4 v0; v0.x = o[0]; v0.y = o[1]; v0.z = o[2]; v0.w = o[3];
  float4 v1; v1.x = o[4]; v1.y = o[5]; v1.z = o[6]; v1.w = o[7];
  dst[0] = v0; dst[1] = v1;
}

// ---------------------------------------------------------------- aux loss
__global__ void aux_kernel(const float* __restrict__ acc, float* __restrict__ out_aux) {
  float mean = 0.f;
  for (int e = 0; e < 8; e++) mean += acc[e];
  mean *= 0.125f;
  float var = 0.f;
  for (int e = 0; e < 8; e++) { float d = acc[e] - mean; var += d * d; }
  var *= 0.125f;
  float il = var / 64.0f;
  float lb = 0.f;
  const float invN = 1.0f / (float)NTOK;
  for (int e = 0; e < 8; e++) lb += (acc[8 + e] * invN) * (acc[16 + e] * invN);
  lb *= 8.0f;
  out_aux[0] = il + lb;
}

// ---------------------------------------------------------------- launch
extern "C" void kernel_launch(void* const* d_in, const int* in_sizes, int n_in,
                              void* d_out, int out_size, void* d_ws, size_t ws_size,
                              hipStream_t stream) {
  const float* x  = (const float*)d_in[0];
  const float* Wg = (const float*)d_in[1];
  const float* bg = (const float*)d_in[2];
  const float* W1 = (const float*)d_in[3];
  const float* b1 = (const float*)d_in[4];
  const float* W2 = (const float*)d_in[5];
  const float* b2 = (const float*)d_in[6];
  float* out = (float*)d_out;

  char* ws = (char*)d_ws;
  size_t off = 0;
  auto carve = [&](size_t bytes) -> void* {
    void* p = ws + off;
    off += (bytes + 255) & ~(size_t)255;
    return p;
  };
  ushort_t* buf  = (ushort_t*)carve((size_t)NEXP * CAP  * DMODEL * 2);
  ushort_t* W1t  = (ushort_t*)carve((size_t)NEXP * HDIM * DMODEL * 2);
  ushort_t* W2t  = (ushort_t*)carve((size_t)NEXP * HDIM * HDIM   * 2);
  ushort_t* Hbuf = (ushort_t*)carve((size_t)NEXP * CAP  * HDIM   * 2);
  float*    Ybuf = (float*)   carve((size_t)NEXP * CAP  * HDIM   * 4);
  int*      tki  = (int*)     carve((size_t)NTOK * TOPK * 4);
  float*    tkp  = (float*)   carve((size_t)NTOK * TOPK * 4);
  int*      rnk  = (int*)     carve((size_t)NTOK * TOPK * 4);
  float*    acc  = (float*)   carve(64 * 4);

  zero_acc_kernel<<<1, 64, 0, stream>>>(acc);
  gating_kernel<<<NTOK / 8, 256, 0, stream>>>(x, Wg, bg, tki, tkp, acc);
  rank_kernel<<<1, NEXP, 0, stream>>>(tki, rnk);
  dispatch_kernel<<<NTOK * TOPK, 256, 0, stream>>>(x, tki, rnk, buf);
  wconv1_kernel<<<(NEXP * DMODEL * HDIM) / 256, 256, 0, stream>>>(W1, W1t);
  wconv2_kernel<<<(NEXP * HDIM * HDIM) / 256, 256, 0, stream>>>(W2, W2t);
  gemm1_kernel<<<dim3(CAP / 128, HDIM / 256, NEXP), 256, 0, stream>>>(buf, W1t, b1, Hbuf);
  gemm2_kernel<<<dim3(CAP / 128, HDIM / 256, NEXP), 256, 0, stream>>>(Hbuf, W2t, b2, Ybuf);
  combine_kernel<<<NTOK, 256, 0, stream>>>(Ybuf, tki, tkp, rnk, out);
  aux_kernel<<<1, 1, 0, stream>>>(acc, out + (size_t)NTOK * HDIM);
}
